// Status_ClassifierHead_67130338836604
// MI455X (gfx1250) — matrix operand order repaired and checked by reading, not compiled, not run on hardware
//
#include <hip/hip_runtime.h>
#include <hip/hip_bf16.h>

// ---------------- problem constants ----------------
#define BB 16
#define NN 1024
#define CC 256
#define NHEAD 8
#define HD 32
#define H0 64
#define W0 80
#define S0 (H0*W0)            // 5120
#define SPOOL 1680            // 1280 + 320 + 80 (levels 1..3)
#define STOT 6800             // 5120 + 1680
#define NCLS 7

typedef __attribute__((ext_vector_type(16))) __bf16 v16bf;
typedef __attribute__((ext_vector_type(8)))  float  v8f;

// =====================================================================
// Weight repack: W (K x N, fp32 row-major) -> bf16 fragment-native tiles.
// Tile (nt, kt) covers cols [nt*16, nt*16+16) and k in [kt*32, kt*32+32).
// Per lane (g = lane/16, li = lane%16): col = nt*16+li, K = kt*32+16g+e,
// 16 bf16 stored contiguously (32 B) so the GEMM B-load is one v16bf.
// Cols >= N are zero-padded.
// =====================================================================
__global__ __launch_bounds__(256)
void repack_w_kernel(const float* __restrict__ W, __bf16* __restrict__ Wp,
                     int K, int N)
{
    const int ntiles = (N + 15) >> 4;
    const int KT = K >> 5;
    int idx = blockIdx.x * 256 + threadIdx.x;       // over ntiles*KT*32 lanes
    if (idx >= ntiles * KT * 32) return;
    int lane = idx & 31;
    int t    = idx >> 5;                            // nt*KT + kt
    int kt   = t % KT, nt = t / KT;
    int g = lane >> 4, li = lane & 15;
    int col = nt * 16 + li;
    int k0  = kt * 32 + 16 * g;
    __bf16* out = Wp + (size_t)idx * 16;
    #pragma unroll
    for (int e = 0; e < 16; ++e) {
        float v = (col < N) ? W[(size_t)(k0 + e) * N + col] : 0.0f;
        out[e] = (__bf16)v;
    }
}

// =====================================================================
// Pyramid pooling: levels 1..3 computed directly from h_t as 2/4/8-block
// means (mean of means == mean over the whole block).
// =====================================================================
__global__ __launch_bounds__(256)
void pool_kernel(const float* __restrict__ h, float* __restrict__ fp)
{
    size_t idx = (size_t)blockIdx.x * 256 + threadIdx.x;
    const size_t total = (size_t)BB * SPOOL * CC;
    if (idx >= total) return;
    int c = (int)(idx & (CC - 1));
    size_t r = idx >> 8;
    int b = (int)(r / SPOOL);
    int s = (int)(r % SPOOL);
    int lvl, Wl, off;
    if (s < 1280)      { lvl = 1; Wl = 40; off = s;        }
    else if (s < 1600) { lvl = 2; Wl = 20; off = s - 1280; }
    else               { lvl = 3; Wl = 10; off = s - 1600; }
    int y = off / Wl, x = off % Wl;
    int blk = 1 << lvl;
    const float* base = h + ((size_t)b * S0) * CC + c;
    float sum = 0.f;
    for (int dy = 0; dy < blk; ++dy)
        for (int dx = 0; dx < blk; ++dx)
            sum += base[(size_t)((y * blk + dy) * W0 + (x * blk + dx)) * CC];
    fp[idx] = sum * (1.0f / (float)(blk * blk));
}

// =====================================================================
// Generic bf16 WMMA GEMM: C[M,N] = act(A[M,K] @ W[K,N] + bias)
// W comes pre-packed (repack_w_kernel). NPER adjacent 16-col tiles are
// computed per wave from one shared A fragment (NPER WMMAs per k-step),
// maximizing A-fragment reuse and independent accumulation chains.
// MODE 0: plain fp32 A (row stride K)
// MODE 1: value mode — rows are B*6800 feature rows, split h_t / fpool
// MODE 2: K-concat — k < K1 from A (stride K1), rest from A2
// =====================================================================
template<int MODE, bool RELU, bool BF16OUT, int NPER>
__global__ __launch_bounds__(256)
void gemm_wmma_bf16(const float* __restrict__ A, const float* __restrict__ A2,
                    const __bf16* __restrict__ Wp, const float* __restrict__ bias,
                    float* __restrict__ Cout, __bf16* __restrict__ CoutBf,
                    int M, int N, int K, int K1)
{
    const int lane   = threadIdx.x & 31;
    const int KT     = K >> 5;
    const int ngroups = ((N + 15) >> 4) / NPER;
    const int wave = blockIdx.x * 8 + (threadIdx.x >> 5);
    const int mt = wave / ngroups;
    const int ng = wave % ngroups;
    if (mt * 16 >= M) return;               // wave-uniform: EXEC stays full
    const int g  = lane >> 4;
    const int li = lane & 15;
    const int m  = mt * 16 + li;            // A row owned by this lane

    const float* arow1 = nullptr;
    const float* arow2 = nullptr;
    if (MODE == 0) {
        arow1 = A + (size_t)m * K;
    } else if (MODE == 1) {
        int b = m / STOT, s = m - b * STOT;
        arow1 = (s < S0) ? (A  + ((size_t)b * S0    + s)        * CC)
                         : (A2 + ((size_t)b * SPOOL + (s - S0)) * CC);
    } else {
        arow1 = A  + (size_t)m * K1;
        arow2 = A2 + (size_t)m * (K - K1);
    }

    v8f acc[NPER];
    #pragma unroll
    for (int t = 0; t < NPER; ++t) acc[t] = (v8f){};

    for (int kt = 0; kt < KT; ++kt) {
        const int k0 = kt * 32;
        const float* ar; int kk;
        if (MODE == 2) {
            if (k0 < K1) { ar = arow1; kk = k0; }
            else         { ar = arow2; kk = k0 - K1; }
        } else { ar = arow1; kk = k0; }

        // A fragment: lane m, K = kk + 16g+e (e<16)
        v16bf afrag;
        #pragma unroll
        for (int e = 0; e < 8; ++e)
            afrag[e] = (__bf16)ar[kk + 16 * g + e];
        #pragma unroll
        for (int e = 0; e < 8; ++e)
            afrag[8 + e] = (__bf16)ar[kk + 16 * g + 8 + e];

        #pragma unroll
        for (int t = 0; t < NPER; ++t) {
            const int nt = ng * NPER + t;
            const v16bf bfrag =
                *(const v16bf*)(Wp + (((size_t)nt * KT + kt) * 32 + lane) * 16);
            acc[t] = __builtin_amdgcn_wmma_f32_16x16x32_bf16(
                         false, afrag, false, bfrag, (short)0, acc[t], false, false);
        }
    }

    #pragma unroll
    for (int t = 0; t < NPER; ++t) {
        const int col = (ng * NPER + t) * 16 + li;
        float bv = (col < N) ? bias[col] : 0.0f;
        #pragma unroll
        for (int r = 0; r < 8; ++r) {
            int row = mt * 16 + r + 8 * g;
            if (col < N) {
                float v = acc[t][r] + bv;
                if (RELU) v = v > 0.f ? v : 0.f;
                if (BF16OUT) CoutBf[(size_t)row * N + col] = (__bf16)v;
                else         Cout  [(size_t)row * N + col] = v;
            }
        }
    }
}

// =====================================================================
// Softmax over the innermost 16 (NLVL*NPTS) of attn logits (B*N*8, 16)
// =====================================================================
__global__ __launch_bounds__(256)
void softmax16_kernel(float* __restrict__ attn, int total)
{
    int i = blockIdx.x * 256 + threadIdx.x;
    if (i >= total) return;
    float* p = attn + (size_t)i * 16;
    float mx = p[0];
    #pragma unroll
    for (int j = 1; j < 16; ++j) mx = fmaxf(mx, p[j]);
    float e[16]; float s = 0.f;
    #pragma unroll
    for (int j = 0; j < 16; ++j) { e[j] = __expf(p[j] - mx); s += e[j]; }
    float inv = 1.0f / s;
    #pragma unroll
    for (int j = 0; j < 16; ++j) p[j] = e[j] * inv;
}

// =====================================================================
// Multi-scale deformable bilinear sampling.
// One wave32 per (b, n, head); lane = channel within head (HD = 32).
// value is bf16 (L2-resident); gathers are 64B coalesced per corner.
// =====================================================================
__device__ __forceinline__
float corner_load(const __bf16* vb, int base, int Wl, int Hl, int xi, int yi)
{
    if (xi < 0 || xi >= Wl || yi < 0 || yi >= Hl) return 0.f;
    return (float)vb[(size_t)(base + yi * Wl + xi) * CC];
}

__global__ __launch_bounds__(256)
void sample_kernel(const __bf16* __restrict__ value,   // (B, 6800, 256) bf16
                   const float* __restrict__ offs,     // (B*N, 256)
                   const float* __restrict__ attn,     // (B*N, 128) softmaxed
                   const float* __restrict__ phead,    // (B*N, 2)
                   float* __restrict__ out)            // (B*N, 256)
{
    const int lane = threadIdx.x & 31;
    int w  = blockIdx.x * 8 + (threadIdx.x >> 5);   // over B*N*NHEAD
    int h  = w & (NHEAD - 1);
    int bn = w >> 3;
    int b  = bn >> 10;                              // N == 1024

    float px = phead[(size_t)bn * 2 + 0] * (1.0f / 640.0f);
    float py = phead[(size_t)bn * 2 + 1] * (1.0f / 512.0f);
    px = fminf(fmaxf(px, 0.f), 1.f);
    py = fminf(fmaxf(py, 0.f), 1.f);

    const float*  orow = offs + (size_t)bn * CC;
    const float*  arow = attn + (size_t)bn * 128 + h * 16;
    const __bf16* vb   = value + ((size_t)b * STOT) * CC + h * HD + lane;

    const int Wls[4]   = {80, 40, 20, 10};
    const int Hls[4]   = {64, 32, 16, 8};
    const int bases[4] = {0, 5120, 6400, 6720};

    float acc = 0.f;
    #pragma unroll
    for (int l = 0; l < 4; ++l) {
        const int   Wl = Wls[l], Hl = Hls[l], base = bases[l];
        const float fw = (float)Wl, fh = (float)Hl;
        #pragma unroll
        for (int p = 0; p < 4; ++p) {
            int oi = ((h * 4 + l) * 4 + p) * 2;
            float lx = px + orow[oi + 0] / fw;
            float ly = py + orow[oi + 1] / fh;
            float x = lx * fw - 0.5f;
            float y = ly * fh - 0.5f;
            float x0f = floorf(x), y0f = floorf(y);
            float wx = x - x0f, wy = y - y0f;
            int x0 = (int)x0f, y0 = (int)y0f;
            float v00 = corner_load(vb, base, Wl, Hl, x0,     y0);
            float v10 = corner_load(vb, base, Wl, Hl, x0 + 1, y0);
            float v01 = corner_load(vb, base, Wl, Hl, x0,     y0 + 1);
            float v11 = corner_load(vb, base, Wl, Hl, x0 + 1, y0 + 1);
            float bil = v00 * (1.f - wx) * (1.f - wy) + v10 * wx * (1.f - wy)
                      + v01 * (1.f - wx) * wy         + v11 * wx * wy;
            acc += arow[l * 4 + p] * bil;
        }
    }
    out[(size_t)bn * CC + h * HD + lane] = acc;
}

// =====================================================================
// Host side
// =====================================================================
extern "C" void kernel_launch(void* const* d_in, const int* in_sizes, int n_in,
                              void* d_out, int out_size, void* d_ws, size_t ws_size,
                              hipStream_t stream)
{
    (void)in_sizes; (void)n_in; (void)out_size; (void)ws_size;
    const float* q_t    = (const float*)d_in[0];
    const float* h_t    = (const float*)d_in[1];
    const float* p_head = (const float*)d_in[2];
    const float* W_off  = (const float*)d_in[3];
    const float* b_off  = (const float*)d_in[4];
    const float* W_attn = (const float*)d_in[5];
    const float* b_attn = (const float*)d_in[6];
    const float* W_v    = (const float*)d_in[7];
    const float* b_v    = (const float*)d_in[8];
    const float* W_o    = (const float*)d_in[9];
    const float* b_o    = (const float*)d_in[10];
    const float* W_p1   = (const float*)d_in[11];
    const float* b_p1   = (const float*)d_in[12];
    const float* W_p2   = (const float*)d_in[13];
    const float* b_p2   = (const float*)d_in[14];
    float* logits = (float*)d_out;

    // -------- workspace layout (with reuse; ~126 MB) --------
    char* ws = (char*)d_ws;
    const size_t SZ_FPOOL = (size_t)BB * SPOOL * CC * 4;   // 27.5 MB fp32
    const size_t SZ_VALUE = (size_t)BB * STOT  * CC * 2;   // 55.7 MB bf16
    const size_t SZ_OFFS  = (size_t)BB * NN    * CC * 4;   // 16.8 MB fp32
    const size_t SZ_ATTN  = (size_t)BB * NN    * 128 * 4;  //  8.4 MB fp32
    const size_t SZ_OUTV  = (size_t)BB * NN    * CC * 4;   // 16.8 MB fp32
    float*  fpool = (float*)(ws);
    __bf16* value = (__bf16*)(ws + SZ_FPOOL);
    float*  offs  = (float*)(ws + SZ_FPOOL + SZ_VALUE);
    float*  attn  = (float*)(ws + SZ_FPOOL + SZ_VALUE + SZ_OFFS);
    float*  outv  = (float*)(ws + SZ_FPOOL + SZ_VALUE + SZ_OFFS + SZ_ATTN);
    char*   wpk   = ws + SZ_FPOOL + SZ_VALUE + SZ_OFFS + SZ_ATTN + SZ_OUTV;
    float*  qloc  = offs;    // offsets dead after sampling
    float*  hid   = fpool;   // pyramid dead after value GEMM (16.8 <= 27.5 MB)

    // packed weights (bf16 elems = ceil(N/16)*16 * K)
    __bf16* Wp_v    = (__bf16*)(wpk);                // 16 tiles * 8kt * 512 = 128 KB
    __bf16* Wp_off  = Wp_v    + 16 * 8  * 32 * 16;
    __bf16* Wp_attn = Wp_off  + 16 * 8  * 32 * 16;   //  8 tiles * 8kt      =  64 KB
    __bf16* Wp_o    = Wp_attn +  8 * 8  * 32 * 16;
    __bf16* Wp_p1   = Wp_o    + 16 * 8  * 32 * 16;   // 16 tiles * 16kt     = 256 KB
    __bf16* Wp_p2   = Wp_p1   + 16 * 16 * 32 * 16;   //  1 tile  * 8kt      =   8 KB

    const int Mq = BB * NN;          // 16384
    const int Mv = BB * STOT;        // 108800

    // 0) repack all weight matrices to bf16 fragment layout (tiny)
    repack_w_kernel<<<dim3(16), dim3(256), 0, stream>>>(W_v,    Wp_v,    256, 256);
    repack_w_kernel<<<dim3(16), dim3(256), 0, stream>>>(W_off,  Wp_off,  256, 256);
    repack_w_kernel<<<dim3(8),  dim3(256), 0, stream>>>(W_attn, Wp_attn, 256, 128);
    repack_w_kernel<<<dim3(16), dim3(256), 0, stream>>>(W_o,    Wp_o,    256, 256);
    repack_w_kernel<<<dim3(32), dim3(256), 0, stream>>>(W_p1,   Wp_p1,   512, 256);
    repack_w_kernel<<<dim3(1),  dim3(256), 0, stream>>>(W_p2,   Wp_p2,   256, NCLS);

    // 1) pyramid pooling (levels 1..3)
    pool_kernel<<<dim3((BB * SPOOL * CC + 255) / 256), dim3(256), 0, stream>>>(h_t, fpool);

    // 2) value = f_scales @ W_v + b_v -> bf16    (108800 x 256 x 256)
    gemm_wmma_bf16<1, false, true, 4><<<dim3(Mv / 16 * 4 / 8), dim3(256), 0, stream>>>(
        h_t, fpool, Wp_v, b_v, nullptr, value, Mv, CC, CC, 0);

    // 3) offsets = q_t @ W_off + b_off           (16384 x 256 x 256)
    gemm_wmma_bf16<0, false, false, 4><<<dim3(Mq / 16 * 4 / 8), dim3(256), 0, stream>>>(
        q_t, nullptr, Wp_off, b_off, offs, nullptr, Mq, CC, CC, 0);

    // 4) attn logits = q_t @ W_attn + b_attn     (16384 x 128 x 256)
    gemm_wmma_bf16<0, false, false, 4><<<dim3(Mq / 16 * 2 / 8), dim3(256), 0, stream>>>(
        q_t, nullptr, Wp_attn, b_attn, attn, nullptr, Mq, 128, CC, 0);

    // 5) softmax over 16 per (b, n, head)
    softmax16_kernel<<<dim3((Mq * NHEAD + 255) / 256), dim3(256), 0, stream>>>(attn, Mq * NHEAD);

    // 6) deformable bilinear sampling -> outv (B*N, 256)
    sample_kernel<<<dim3(Mq * NHEAD / 8), dim3(256), 0, stream>>>(value, offs, attn, p_head, outv);

    // 7) q_local = outv @ W_o + b_o              (16384 x 256 x 256)
    gemm_wmma_bf16<0, false, false, 4><<<dim3(Mq / 16 * 4 / 8), dim3(256), 0, stream>>>(
        outv, nullptr, Wp_o, b_o, qloc, nullptr, Mq, CC, CC, 0);

    // 8) hid = relu([q_t, q_local] @ W_p1 + b_p1)  (16384 x 256 x 512, K-concat)
    gemm_wmma_bf16<2, true, false, 4><<<dim3(Mq / 16 * 4 / 8), dim3(256), 0, stream>>>(
        q_t, qloc, Wp_p1, b_p1, hid, nullptr, Mq, CC, 2 * CC, CC);

    // 9) logits = hid @ W_p2 + b_p2              (16384 x 7 x 256, cols padded)
    gemm_wmma_bf16<0, false, false, 1><<<dim3(Mq / 16 * 1 / 8), dim3(256), 0, stream>>>(
        hid, nullptr, Wp_p2, b_p2, logits, nullptr, Mq, NCLS, CC, 0);
}
